// Net_73710228734901
// MI455X (gfx1250) — matrix operand order repaired and checked by reading, not compiled, not run on hardware
//
#include <hip/hip_runtime.h>
#include <hip/hip_bf16.h>

typedef __attribute__((ext_vector_type(16))) _Float16 v16h;
typedef __attribute__((ext_vector_type(8)))  _Float16 v8h;
typedef __attribute__((ext_vector_type(8)))  float    v8f;

#define B_ 64
#define CH3 1280
#define CH  1024
#define HW  196          // 14*14
#define K1  1280         // GEMM K
#define NT2 7            // ceil(196/32) hw super-tiles (32 wide)
#define MT2 32           // 1024/32 o super-tiles (32 tall)
#define NTP 13           // ceil(196/16) hw tiles (pool kernel)

// ---------------- workspace layout (bytes) ----------------
// p16T : [B][196][1280] f16   (pooled, transposed so K is contiguous for WMMA B)
// cw16 : [1024][1280] f16
// x3p  : [B][1024][196] f32   (1x1-conv output)
// rect : [B][1024][9]  f32    (rectangular plane sums)
// eff  : [1024*9]      f32    (folded conv3d x linear weights / 196)
static const size_t OFF_P16  = 0;
static const size_t OFF_CW16 = OFF_P16  + (size_t)B_*HW*CH3*2;
static const size_t OFF_X3P  = OFF_CW16 + (size_t)CH*K1*2;
static const size_t OFF_RECT = OFF_X3P  + (size_t)B_*CH*HW*4;
static const size_t OFF_EFF  = OFF_RECT + (size_t)B_*CH*9*4;

// --------- kernel 1: avg_pool2d(5,2,2) + transpose + f16 cast ----------
__global__ void k_pool(const float* __restrict__ x3, _Float16* __restrict__ p16T) {
  int blk = blockIdx.x;
  int hwt = blk % NTP;
  int cht = (blk / NTP) % 80;
  int b   = blk / (NTP * 80);
  int t   = threadIdx.x;
  __shared__ float tile[16][17];

  int lc = t >> 4, lhw = t & 15;
  int ch = cht * 16 + lc;
  int hw = hwt * 16 + lhw;
  float s = 0.f;
  if (hw < HW) {
    int h = hw / 14, w = hw - (hw / 14) * 14;
    const float* src = x3 + ((size_t)b * CH3 + ch) * 784;
    #pragma unroll
    for (int dh = -2; dh <= 2; ++dh) {
      int r = 2 * h + dh;
      if (r < 0 || r > 27) continue;
      #pragma unroll
      for (int dw = -2; dw <= 2; ++dw) {
        int cc = 2 * w + dw;
        if (cc < 0 || cc > 27) continue;
        s += src[r * 28 + cc];
      }
    }
    s *= (1.0f / 25.0f);
  }
  tile[lc][lhw] = s;
  __syncthreads();

  int whw = hwt * 16 + (t >> 4);
  int wch = cht * 16 + (t & 15);
  if (whw < HW)
    p16T[((size_t)b * HW + whw) * CH3 + wch] = (_Float16)tile[t & 15][t >> 4];
}

// --------- kernel 2: cast c_w -> f16 ----------
__global__ void k_cast(const float* __restrict__ cw, _Float16* __restrict__ cw16, int n) {
  int i = blockIdx.x * blockDim.x + threadIdx.x;
  if (i < n) cw16[i] = (_Float16)cw[i];
}

// --------- kernel 3: 1x1 conv as WMMA f16 GEMM, 2x2 register-tiled ----------
// out[b,o,hw] = sum_c cw[o,c] * pool[b,c,hw]
// one wave computes a 32(o) x 32(hw) block = 4 accumulators; 4 wmma per 8 b128 loads
__global__ void k_gemm(const _Float16* __restrict__ cw16,
                       const _Float16* __restrict__ p16T,
                       float* __restrict__ x3p) {
  int wave = blockIdx.x * 8 + (threadIdx.x >> 5);
  int lane = threadIdx.x & 31;
  int nt = wave % NT2;
  int mt = (wave / NT2) & 31;
  int b  = wave / (NT2 * MT2);
  int hs  = lane >> 4;
  int l16 = lane & 15;

  int m0 = mt * 32 + l16;                // A rows (o)
  int n0 = nt * 32 + l16;                // B rows (hw)
  int n1 = n0 + 16;
  int n0c = n0 > (HW - 1) ? (HW - 1) : n0;
  int n1c = n1 > (HW - 1) ? (HW - 1) : n1;

  const _Float16* __restrict__ Ar0 = cw16 + (size_t)m0 * K1;
  const _Float16* __restrict__ Ar1 = Ar0 + (size_t)16 * K1;
  const _Float16* __restrict__ Br0 = p16T + ((size_t)b * HW + n0c) * K1;
  const _Float16* __restrict__ Br1 = p16T + ((size_t)b * HW + n1c) * K1;

  v8f acc00 = {}, acc01 = {}, acc10 = {}, acc11 = {};
  #pragma unroll 2
  for (int k0 = 0; k0 < K1; k0 += 32) {
    int ka = k0 + hs * 16;               // A: elems 0..15 = K[ka..ka+15]
    v8h a0lo = *(const v8h*)(Ar0 + ka);
    v8h a0hi = *(const v8h*)(Ar0 + ka + 8);
    v8h a1lo = *(const v8h*)(Ar1 + ka);
    v8h a1hi = *(const v8h*)(Ar1 + ka + 8);
    v16h a0, a1;
    #pragma unroll
    for (int q = 0; q < 8; ++q) {
      a0[q] = a0lo[q]; a0[q + 8] = a0hi[q];
      a1[q] = a1lo[q]; a1[q + 8] = a1hi[q];
    }
    int kb = k0 + hs * 16;               // B: lanes 0-15 K=k0..k0+15, lanes 16-31 +16
    v16h b0 = *(const v16h*)(Br0 + kb);
    v16h b1 = *(const v16h*)(Br1 + kb);
    acc00 = __builtin_amdgcn_wmma_f32_16x16x32_f16(false, a0, false, b0, (short)0, acc00, false, false);
    acc01 = __builtin_amdgcn_wmma_f32_16x16x32_f16(false, a0, false, b1, (short)0, acc01, false, false);
    acc10 = __builtin_amdgcn_wmma_f32_16x16x32_f16(false, a1, false, b0, (short)0, acc10, false, false);
    acc11 = __builtin_amdgcn_wmma_f32_16x16x32_f16(false, a1, false, b1, (short)0, acc11, false, false);
  }

  // D layout: VGPR v -> M = v + 8*half_sel, N = lane&15
  #pragma unroll
  for (int v = 0; v < 8; ++v) {
    int o0 = mt * 32 + hs * 8 + v;
    int o1 = o0 + 16;
    if (n0 < HW) {
      x3p[((size_t)b * CH + o0) * HW + n0] = acc00[v];
      x3p[((size_t)b * CH + o1) * HW + n0] = acc10[v];
    }
    if (n1 < HW) {
      x3p[((size_t)b * CH + o0) * HW + n1] = acc01[v];
      x3p[((size_t)b * CH + o1) * HW + n1] = acc11[v];
    }
  }
}

// --------- kernel 4: fused genetic-gather + 9 rectangular plane sums ----------
__global__ void k_rect(const float* __restrict__ x1, const float* __restrict__ x2,
                       const float* __restrict__ x3p, const float* __restrict__ xs,
                       const int* __restrict__ idxh, const int* __restrict__ idxw,
                       float* __restrict__ rect) {
  int wave = blockIdx.x * 8 + (threadIdx.x >> 5);
  int lane = threadIdx.x & 31;
  int c = wave & (CH - 1);
  int b = wave >> 10;
  int ih0 = idxh[c],        iw0 = idxw[c];
  int ih1 = idxh[CH + c],   iw1 = idxw[CH + c];
  int ih2 = idxh[2*CH + c], iw2 = idxw[2*CH + c];
  int ih3 = idxh[3*CH + c], iw3 = idxw[3*CH + c];
  size_t base = ((size_t)b * CH + c) * HW;

  float T=0.f, r0=0.f, r13=0.f, c0s=0.f, c13s=0.f, q00=0.f, q0e=0.f, qe0=0.f, qee=0.f;
  for (int e = lane; e < HW; e += 32) {
    int i = e / 14, j = e - (e / 14) * 14;
    float v;
    if (j < 7) {
      if (i < 7) v = x1[base + (ih0 + i)     * 14 + (iw0 + j)];
      else       v = x2[base + (ih1 + i - 7) * 14 + (iw1 + j)];
    } else {
      if (i < 7) v = x3p[base + (ih2 + i)     * 14 + (iw2 + j - 7)];
      else       v = xs [base + (ih3 + i - 7) * 14 + (iw3 + j - 7)];
    }
    T += v;
    if (i == 0)  { r0  += v; if (j == 0) q00 = v; if (j == 13) q0e = v; }
    if (i == 13) { r13 += v; if (j == 0) qe0 = v; if (j == 13) qee = v; }
    if (j == 0)  c0s  += v;
    if (j == 13) c13s += v;
  }
  #pragma unroll
  for (int off = 16; off > 0; off >>= 1) {
    T    += __shfl_xor(T,    off, 32);
    r0   += __shfl_xor(r0,   off, 32);
    r13  += __shfl_xor(r13,  off, 32);
    c0s  += __shfl_xor(c0s,  off, 32);
    c13s += __shfl_xor(c13s, off, 32);
    q00  += __shfl_xor(q00,  off, 32);
    q0e  += __shfl_xor(q0e,  off, 32);
    qe0  += __shfl_xor(qe0,  off, 32);
    qee  += __shfl_xor(qee,  off, 32);
  }
  if (lane == 0) {
    float* R = rect + (size_t)wave * 9;   // wave == b*1024 + c
    R[0] = T - r13 - c13s + qee;
    R[1] = T - r13;
    R[2] = T - r13 - c0s + qe0;
    R[3] = T - c13s;
    R[4] = T;
    R[5] = T - c0s;
    R[6] = T - r0 - c13s + q0e;
    R[7] = T - r0;
    R[8] = T - r0 - c0s + q00;
  }
}

// --------- kernel 5: fold conv3d weights x linear head into eff[c][khw] ----------
__global__ void k_eff(const float* __restrict__ w3, const float* __restrict__ lin_w,
                      float* __restrict__ eff) {
  int idx = blockIdx.x * blockDim.x + threadIdx.x;
  if (idx >= CH * 9) return;
  int khw = idx % 9;
  int c   = idx / 9;
  int ic = c >> 6, zd = c & 63;
  int kd = (zd + 4) % 3;
  int d  = (zd + 4 - kd) / 3;
  float s = 0.f;
  #pragma unroll
  for (int oc = 0; oc < 32; ++oc)
    s += lin_w[oc * 24 + d] * w3[((oc * 16 + ic) * 3 + kd) * 9 + khw];
  eff[idx] = s * (1.0f / 196.0f);
}

// --------- kernel 6: per-batch contraction + bias ----------
__global__ void k_final(const float* __restrict__ rect, const float* __restrict__ eff,
                        const float* __restrict__ lin_b, float* __restrict__ out) {
  int b = blockIdx.x, t = threadIdx.x;
  __shared__ float sm[256];
  const float* R = rect + (size_t)b * (CH * 9);
  float s = 0.f;
  for (int i = t; i < CH * 9; i += 256) s += R[i] * eff[i];
  sm[t] = s;
  __syncthreads();
  for (int off = 128; off > 0; off >>= 1) {
    if (t < off) sm[t] += sm[t + off];
    __syncthreads();
  }
  if (t == 0) out[b] = sm[0] + lin_b[0];
}

extern "C" void kernel_launch(void* const* d_in, const int* in_sizes, int n_in,
                              void* d_out, int out_size, void* d_ws, size_t ws_size,
                              hipStream_t stream) {
  const float* x1    = (const float*)d_in[0];
  const float* x2    = (const float*)d_in[1];
  const float* x3    = (const float*)d_in[2];
  const float* xs    = (const float*)d_in[3];
  const float* c_w   = (const float*)d_in[4];
  const float* w3    = (const float*)d_in[5];
  const float* lin_w = (const float*)d_in[6];
  const float* lin_b = (const float*)d_in[7];
  const int*   idxh  = (const int*)d_in[8];
  const int*   idxw  = (const int*)d_in[9];
  float* out = (float*)d_out;

  char* ws = (char*)d_ws;
  _Float16* p16T = (_Float16*)(ws + OFF_P16);
  _Float16* cw16 = (_Float16*)(ws + OFF_CW16);
  float*    x3p  = (float*)(ws + OFF_X3P);
  float*    rect = (float*)(ws + OFF_RECT);
  float*    eff  = (float*)(ws + OFF_EFF);

  // 1) avg_pool(5,2,2) -> f16, transposed [b][hw][c]
  k_pool<<<B_ * 80 * NTP, 256, 0, stream>>>(x3, p16T);
  // 2) c_w -> f16
  k_cast<<<(CH * K1 + 255) / 256, 256, 0, stream>>>(c_w, cw16, CH * K1);
  // 3) 1x1 conv GEMM via v_wmma_f32_16x16x32_f16 (2x2 register tiles)
  k_gemm<<<(B_ * MT2 * NT2) / 8, 256, 0, stream>>>(cw16, p16T, x3p);
  // 4) fused gather + rectangular sums
  k_rect<<<(B_ * CH) / 8, 256, 0, stream>>>(x1, x2, x3p, xs, idxh, idxw, rect);
  // 5) fold conv3d + linear weights
  k_eff<<<(CH * 9 + 255) / 256, 256, 0, stream>>>(w3, lin_w, eff);
  // 6) final contraction
  k_final<<<B_, 256, 0, stream>>>(rect, eff, lin_b, out);
}
